// DeformableTransformerDecoderLayer_19301583028377
// MI455X (gfx1250) — matrix operand order repaired and checked by reading, not compiled, not run on hardware
//
#include <hip/hip_runtime.h>
#include <math.h>

// ---------------------------------------------------------------------------
// DeformableTransformerDecoderLayer for MI455X (gfx1250, wave32, WMMA).
// All dense GEMMs (X @ W^T + b [+res][+relu]) go through
// v_wmma_f32_16x16x32_f16. Operands are converted fp32->f16 ONCE into
// workspace, so the GEMM inner loop is pure b128 f16 loads + WMMA
// (memory-bound optimal at 23.3 TB/s). Softmax / LayerNorm / bilinear
// sampling stay in fp32 VALU.
// ---------------------------------------------------------------------------

#define DMODEL 256
#define NHEAD 8
#define HDIM 32
#define NLVL 4
#define NPTS 4
#define DFFN 1024
#define BS 8
#define NQ 900
#define NV 21760
#define ROWS (BS * NQ)            // 7200

typedef _Float16 v16h __attribute__((ext_vector_type(16)));
typedef _Float16 v8h  __attribute__((ext_vector_type(8)));
typedef _Float16 v4h  __attribute__((ext_vector_type(4)));
typedef float    v8f  __attribute__((ext_vector_type(8)));

// ---------------------------------------------------------------------------
// fp32 -> f16 conversion, 4 elems/thread (n must be a multiple of 4)
// ---------------------------------------------------------------------------
__global__ __launch_bounds__(256) void k_cvt(const float4* __restrict__ in,
                                             v4h* __restrict__ out, int n4) {
  int i = blockIdx.x * 256 + threadIdx.x;
  if (i < n4) {
    float4 f = in[i];
    v4h h = {(_Float16)f.x, (_Float16)f.y, (_Float16)f.z, (_Float16)f.w};
    out[i] = h;
  }
}

// ---------------------------------------------------------------------------
// f16 add: oh = (f16)(a + b), 4 elems/thread
// ---------------------------------------------------------------------------
__global__ __launch_bounds__(256) void k_add_h(const float4* __restrict__ a,
                                               const float4* __restrict__ b,
                                               v4h* __restrict__ oh, int n4) {
  int i = blockIdx.x * 256 + threadIdx.x;
  if (i < n4) {
    float4 x = a[i], y = b[i];
    v4h h = {(_Float16)(x.x + y.x), (_Float16)(x.y + y.y),
             (_Float16)(x.z + y.z), (_Float16)(x.w + y.w)};
    oh[i] = h;
  }
}

// ---------------------------------------------------------------------------
// WMMA GEMM: C[M,N] = act(A[M,K] @ W[N,K]^T + bias[N]) (+ resid[M,N])
// A, W row-major f16; C fp32 and/or f16 (either pointer may be null).
// Block = 256 threads = 8 waves; wave tile 32x32; block tile 128(M) x 64(N).
// Requires: K % 32 == 0, N % 64 == 0 (true for all call sites).
// ---------------------------------------------------------------------------
__device__ inline void store_ctile(const v8f& c, int m0, int n0, int M, int N,
                                   const float* __restrict__ bias,
                                   const float* __restrict__ resid,
                                   float* __restrict__ C,
                                   _Float16* __restrict__ Ch,
                                   int relu, int lane) {
  int col = n0 + (lane & 15);
  int rbase = m0 + ((lane >> 4) ? 8 : 0);
  float bv = bias[col];
#pragma unroll
  for (int e = 0; e < 8; ++e) {
    int row = rbase + e;
    if (row < M) {
      float val = c[e] + bv;
      if (relu) val = fmaxf(val, 0.0f);
      if (resid) val += resid[(size_t)row * N + col];
      if (C)  C[(size_t)row * N + col] = val;
      if (Ch) Ch[(size_t)row * N + col] = (_Float16)val;
    }
  }
}

__global__ __launch_bounds__(256) void k_gemm(const _Float16* __restrict__ A,
                                              const _Float16* __restrict__ W,
                                              const float* __restrict__ bias,
                                              const float* __restrict__ resid,
                                              float* __restrict__ C,
                                              _Float16* __restrict__ Ch,
                                              int M, int N, int K, int relu) {
  const int lane = threadIdx.x & 31;
  const int wid  = threadIdx.x >> 5;       // 0..7
  const int wm   = wid & 3;                // 4 waves along M
  const int wn   = wid >> 2;               // 2 waves along N
  const int mblk = blockIdx.y * 128 + wm * 32;
  const int nblk = blockIdx.x * 64  + wn * 32;
  const int hs   = lane >> 4;              // lane half select
  const int l15  = lane & 15;

  // A-frag rows (clamped; garbage rows are masked at store time)
  int mr0 = mblk + l15;       if (mr0 >= M) mr0 = M - 1;
  int mr1 = mblk + 16 + l15;  if (mr1 >= M) mr1 = M - 1;
  const int nc0 = nblk + l15;
  const int nc1 = nblk + 16 + l15;

  const _Float16* __restrict__ Ap0 = A + (size_t)mr0 * K + hs * 16;
  const _Float16* __restrict__ Ap1 = A + (size_t)mr1 * K + hs * 16;
  const _Float16* __restrict__ Wp0 = W + (size_t)nc0 * K + hs * 16;
  const _Float16* __restrict__ Wp1 = W + (size_t)nc1 * K + hs * 16;

  v8f c00 = {}, c01 = {}, c10 = {}, c11 = {};

  for (int k0 = 0; k0 < K; k0 += 32) {
    // A 16x32 frag: elems 0..15 <- K = hs*16 + 0..15
    v8h a0lo = *(const v8h*)(Ap0 + k0);
    v8h a0hi = *(const v8h*)(Ap0 + k0 + 8);
    v8h a1lo = *(const v8h*)(Ap1 + k0);
    v8h a1hi = *(const v8h*)(Ap1 + k0 + 8);
    // B 32x16 frag: lane = col n, elems 0..15 <- K = hs*16 + 0..15 (W row-major)
    v8h b0lo = *(const v8h*)(Wp0 + k0);
    v8h b0hi = *(const v8h*)(Wp0 + k0 + 8);
    v8h b1lo = *(const v8h*)(Wp1 + k0);
    v8h b1hi = *(const v8h*)(Wp1 + k0 + 8);

    // prefetch next K tile (global_prefetch_b8; speculative, OOB-safe)
    __builtin_prefetch(Ap0 + k0 + 32, 0, 1);
    __builtin_prefetch(Wp0 + k0 + 32, 0, 1);

    v16h a0 = __builtin_shufflevector(a0lo, a0hi, 0,1,2,3,4,5,6,7,8,9,10,11,12,13,14,15);
    v16h a1 = __builtin_shufflevector(a1lo, a1hi, 0,1,2,3,4,5,6,7,8,9,10,11,12,13,14,15);
    v16h b0 = __builtin_shufflevector(b0lo, b0hi, 0,1,2,3,4,5,6,7,8,9,10,11,12,13,14,15);
    v16h b1 = __builtin_shufflevector(b1lo, b1hi, 0,1,2,3,4,5,6,7,8,9,10,11,12,13,14,15);

    c00 = __builtin_amdgcn_wmma_f32_16x16x32_f16(false, a0, false, b0, (short)0, c00, false, false);
    c01 = __builtin_amdgcn_wmma_f32_16x16x32_f16(false, a0, false, b1, (short)0, c01, false, false);
    c10 = __builtin_amdgcn_wmma_f32_16x16x32_f16(false, a1, false, b0, (short)0, c10, false, false);
    c11 = __builtin_amdgcn_wmma_f32_16x16x32_f16(false, a1, false, b1, (short)0, c11, false, false);
  }

  store_ctile(c00, mblk,      nblk,      M, N, bias, resid, C, Ch, relu, lane);
  store_ctile(c01, mblk,      nblk + 16, M, N, bias, resid, C, Ch, relu, lane);
  store_ctile(c10, mblk + 16, nblk,      M, N, bias, resid, C, Ch, relu, lane);
  store_ctile(c11, mblk + 16, nblk + 16, M, N, bias, resid, C, Ch, relu, lane);
}

// ---------------------------------------------------------------------------
// Self-attention (fp32, online softmax). 1 wave per (b, h, query).
// q,k,v layout: [b*NQ + row, h*HDIM + d]. out (f16) same layout.
// ---------------------------------------------------------------------------
__global__ __launch_bounds__(256) void k_attn(const float* __restrict__ q,
                                              const float* __restrict__ k,
                                              const float* __restrict__ v,
                                              _Float16* __restrict__ out) {
  const int lane = threadIdx.x & 31;
  const int wid  = threadIdx.x >> 5;
  const int gw   = blockIdx.x * 8 + wid;          // BS*H*NQ = 57600 waves
  const int qi   = gw % NQ;
  const int bh   = gw / NQ;
  const int h    = bh & (NHEAD - 1);
  const int b    = bh >> 3;
  const float scale = 0.1767766952966369f;        // 1/sqrt(32)

  const size_t qoff = ((size_t)(b * NQ + qi)) * DMODEL + h * HDIM;
  const float* __restrict__ qp = q + qoff;
  float qv[HDIM];
#pragma unroll
  for (int d = 0; d < HDIM; ++d) qv[d] = qp[d];

  float mrun = -1e30f, denom = 0.0f, acc = 0.0f;   // acc: lane owns dim d=lane

  for (int kc = 0; kc < NQ; kc += 32) {
    const int j = kc + lane;
    float s = -1e30f;
    if (j < NQ) {
      const float* __restrict__ kp = k + ((size_t)(b * NQ + j)) * DMODEL + h * HDIM;
      float dot = 0.0f;
#pragma unroll
      for (int d = 0; d < HDIM; ++d) dot += qv[d] * kp[d];
      s = dot * scale;
    }
    float cmax = s;
#pragma unroll
    for (int o = 16; o > 0; o >>= 1) cmax = fmaxf(cmax, __shfl_xor(cmax, o, 32));
    const float mnew = fmaxf(mrun, cmax);
    const float corr = __expf(mrun - mnew);
    float p = (j < NQ) ? __expf(s - mnew) : 0.0f;
    float psum = p;
#pragma unroll
    for (int o = 16; o > 0; o >>= 1) psum += __shfl_xor(psum, o, 32);
    denom = denom * corr + psum;
    acc *= corr;
    mrun = mnew;
#pragma unroll 8
    for (int jj = 0; jj < 32; ++jj) {
      const int kj = kc + jj;
      if (kj < NQ) {
        const float pj = __shfl(p, jj, 32);
        acc += pj * v[((size_t)(b * NQ + kj)) * DMODEL + h * HDIM + lane];
      }
    }
  }
  out[qoff + lane] = (_Float16)(acc / denom);
}

// ---------------------------------------------------------------------------
// Residual + LayerNorm: out[row] = LN(a[row] + b[row]) * g + beta.
// Optional f16 copy of the output. 1 block (256 thr) per row of 256.
// ---------------------------------------------------------------------------
__global__ __launch_bounds__(256) void k_res_ln(const float* __restrict__ a,
                                                const float* __restrict__ bb,
                                                const float* __restrict__ g,
                                                const float* __restrict__ beta,
                                                float* __restrict__ out,
                                                _Float16* __restrict__ out_h) {
  const int row = blockIdx.x;
  const int t = threadIdx.x;
  const int lane = t & 31, wid = t >> 5;
  __shared__ float s1[8];
  __shared__ float s2[8];

  const size_t off = (size_t)row * DMODEL + t;
  float x = a[off] + bb[off];

  float s = x;
#pragma unroll
  for (int o = 16; o > 0; o >>= 1) s += __shfl_xor(s, o, 32);
  if (lane == 0) s1[wid] = s;
  __syncthreads();
  float tot = 0.0f;
#pragma unroll
  for (int i = 0; i < 8; ++i) tot += s1[i];
  const float mean = tot * (1.0f / DMODEL);

  const float d = x - mean;
  float vs = d * d;
#pragma unroll
  for (int o = 16; o > 0; o >>= 1) vs += __shfl_xor(vs, o, 32);
  if (lane == 0) s2[wid] = vs;
  __syncthreads();
  float vtot = 0.0f;
#pragma unroll
  for (int i = 0; i < 8; ++i) vtot += s2[i];
  const float var = vtot * (1.0f / DMODEL);

  const float y = d * rsqrtf(var + 1e-5f) * g[t] + beta[t];
  out[off] = y;
  if (out_h) out_h[off] = (_Float16)y;
}

// ---------------------------------------------------------------------------
// MSDA core: 1 wave per (b, q, h); lane owns output dim d.
// off_raw: [row, h*32 + (l*P+p)*2 + c]; aw_raw: [row*128 + h*16 + i]
// ref: [b,q,L,2]; value: [b, NV, H*HDIM] fp32; out (f16): [row, h*HDIM + d]
// Level shapes hardcoded from the reference: {128,64,32,16}^2.
// ---------------------------------------------------------------------------
__global__ __launch_bounds__(256) void k_msda(const float* __restrict__ off_raw,
                                              const float* __restrict__ aw_raw,
                                              const float* __restrict__ ref,
                                              const float* __restrict__ value,
                                              _Float16* __restrict__ out) {
  const int lane = threadIdx.x & 31;
  const int wid  = threadIdx.x >> 5;
  const int gw   = blockIdx.x * 8 + wid;           // BS*NQ*H waves
  const int h    = gw & (NHEAD - 1);
  const int q    = (gw >> 3) % NQ;
  const int b    = gw / (NQ * NHEAD);
  const size_t row = (size_t)(b * NQ + q);

  const int lvl_w[NLVL]  = {128, 64, 32, 16};
  const int lvl_st[NLVL] = {0, 16384, 20480, 21504};

  // softmax over the 16 attention weights of this head
  const float* __restrict__ awp = aw_raw + row * (NHEAD * NLVL * NPTS) + h * 16;
  float aval = (lane < 16) ? awp[lane] : -1e30f;
  float amax = aval;
#pragma unroll
  for (int o = 16; o > 0; o >>= 1) amax = fmaxf(amax, __shfl_xor(amax, o, 32));
  float ae = (lane < 16) ? __expf(aval - amax) : 0.0f;
  float asum = ae;
#pragma unroll
  for (int o = 16; o > 0; o >>= 1) asum += __shfl_xor(asum, o, 32);
  const float ainv = 1.0f / asum;

  const float* __restrict__ offp = off_raw + row * DMODEL + h * (NLVL * NPTS * 2);
  const float* __restrict__ refp = ref + (row * NLVL) * 2;
  float acc = 0.0f;

  for (int l = 0; l < NLVL; ++l) {
    const int Wl = lvl_w[l];
    const float fw = (float)Wl;
    const int st = lvl_st[l];
    const float rx = refp[l * 2 + 0];
    const float ry = refp[l * 2 + 1];
    for (int pp = 0; pp < NPTS; ++pp) {
      const int i = l * NPTS + pp;
      const float pi = __shfl(ae, i, 32) * ainv;
      const float ox = offp[i * 2 + 0];
      const float oy = offp[i * 2 + 1];
      const float lx = rx * fw + ox - 0.5f;   // norm is (W,H), square levels
      const float ly = ry * fw + oy - 0.5f;
      const float x0f = floorf(lx), y0f = floorf(ly);
      const float tx = lx - x0f, ty = ly - y0f;
      const int x0 = (int)x0f, y0 = (int)y0f;
#pragma unroll
      for (int dy = 0; dy < 2; ++dy) {
#pragma unroll
        for (int dx = 0; dx < 2; ++dx) {
          const int xi = x0 + dx, yi = y0 + dy;
          if (xi >= 0 && xi < Wl && yi >= 0 && yi < Wl) {
            const float wgt = (dx ? tx : 1.0f - tx) * (dy ? ty : 1.0f - ty);
            const int idx = st + yi * Wl + xi;
            const float vv = value[(((size_t)b * NV + idx) * DMODEL) + h * HDIM + lane];
            acc += pi * wgt * vv;
          }
        }
      }
    }
  }
  out[row * DMODEL + h * HDIM + lane] = (_Float16)acc;
}

// ---------------------------------------------------------------------------
// launch
// ---------------------------------------------------------------------------
extern "C" void kernel_launch(void* const* d_in, const int* in_sizes, int n_in,
                              void* d_out, int out_size, void* d_ws, size_t ws_size,
                              hipStream_t stream) {
  const float* tgt      = (const float*)d_in[0];
  const float* qpos     = (const float*)d_in[1];
  const float* ref      = (const float*)d_in[2];
  const float* src      = (const float*)d_in[3];
  const float* sa_in_w  = (const float*)d_in[4];
  const float* sa_in_b  = (const float*)d_in[5];
  const float* sa_out_w = (const float*)d_in[6];
  const float* sa_out_b = (const float*)d_in[7];
  const float* ln2_g    = (const float*)d_in[8];
  const float* ln2_b    = (const float*)d_in[9];
  const float* off_w    = (const float*)d_in[10];
  const float* off_b    = (const float*)d_in[11];
  const float* aw_w     = (const float*)d_in[12];
  const float* aw_b     = (const float*)d_in[13];
  const float* val_w    = (const float*)d_in[14];
  const float* val_b    = (const float*)d_in[15];
  const float* out_w    = (const float*)d_in[16];
  const float* out_b    = (const float*)d_in[17];
  const float* ln1_g    = (const float*)d_in[18];
  const float* ln1_b    = (const float*)d_in[19];
  const float* ffn1_w   = (const float*)d_in[20];
  const float* ffn1_b   = (const float*)d_in[21];
  const float* ffn2_w   = (const float*)d_in[22];
  const float* ffn2_b   = (const float*)d_in[23];
  const float* ln3_g    = (const float*)d_in[24];
  const float* ln3_b    = (const float*)d_in[25];

  const size_t R  = (size_t)ROWS * DMODEL;   // 1,843,200
  const size_t SV = (size_t)BS * NV * DMODEL;// 44,564,480

  // workspace carve (bump allocator, everything naturally 16B-aligned)
  char* wp = (char*)d_ws;
  auto allocF = [&](size_t n) { float* r = (float*)wp; wp += n * sizeof(float); return r; };
  auto allocH = [&](size_t n) { _Float16* r = (_Float16*)wp; wp += n * sizeof(_Float16); return r; };

  float* qb    = allocF(R);
  float* kb    = allocF(R);
  float* vb    = allocF(R);
  float* sa2   = allocF(R);
  float* tgt2  = allocF(R);
  float* offr  = allocF(R);
  float* awr   = allocF((size_t)ROWS * 128);
  float* ca    = allocF(R);
  float* tgt3  = allocF(R);
  float* ffno  = allocF(R);
  float* value = allocF(SV);

  _Float16* qk_h     = allocH(R);      // qk, then reused for qk2
  _Float16* tgt_h    = allocH(R);
  _Float16* sa_h     = allocH(R);
  _Float16* msda_h   = allocH(R);
  _Float16* tgt3_h   = allocH(R);
  _Float16* ffnh_h   = allocH((size_t)ROWS * DFFN);
  _Float16* src_h    = allocH(SV);
  _Float16* sa_in_wh = allocH(3 * 256 * 256);
  _Float16* sa_out_wh= allocH(256 * 256);
  _Float16* off_wh   = allocH(256 * 256);
  _Float16* aw_wh    = allocH(128 * 256);
  _Float16* val_wh   = allocH(256 * 256);
  _Float16* out_wh   = allocH(256 * 256);
  _Float16* ffn1_wh  = allocH(1024 * 256);
  _Float16* ffn2_wh  = allocH(256 * 1024);

  const dim3 blk(256);
  const dim3 g256(4, (ROWS + 127) / 128);          // N=256 GEMMs on ROWS
  const dim3 g128(2, (ROWS + 127) / 128);          // N=128
  const dim3 g1024(16, (ROWS + 127) / 128);        // N=1024
  const dim3 gval(4, (BS * NV) / 128);             // value proj, M=174080

  auto cvt = [&](const float* in, _Float16* out, size_t n) {
    int n4 = (int)(n / 4);
    k_cvt<<<(n4 + 255) / 256, blk, 0, stream>>>((const float4*)in, (v4h*)out, n4);
  };

  // 0) one-time f16 conversions of weights + streamed operands
  cvt(sa_in_w,  sa_in_wh,  3 * 256 * 256);
  cvt(sa_out_w, sa_out_wh, 256 * 256);
  cvt(off_w,    off_wh,    256 * 256);
  cvt(aw_w,     aw_wh,     128 * 256);
  cvt(val_w,    val_wh,    256 * 256);
  cvt(out_w,    out_wh,    256 * 256);
  cvt(ffn1_w,   ffn1_wh,   1024 * 256);
  cvt(ffn2_w,   ffn2_wh,   256 * 1024);
  cvt(tgt,      tgt_h,     R);
  cvt(src,      src_h,     SV);

  // 1) qk = tgt + query_pos (f16, GEMM operand only)
  k_add_h<<<(int)(R / 4 / 256), blk, 0, stream>>>((const float4*)tgt, (const float4*)qpos, (v4h*)qk_h, (int)(R / 4));
  // 2-4) q, k (from qk), v (from tgt)
  k_gemm<<<g256, blk, 0, stream>>>(qk_h,  sa_in_wh,             sa_in_b,       nullptr, qb, nullptr, ROWS, 256, 256, 0);
  k_gemm<<<g256, blk, 0, stream>>>(qk_h,  sa_in_wh + 256 * 256, sa_in_b + 256, nullptr, kb, nullptr, ROWS, 256, 256, 0);
  k_gemm<<<g256, blk, 0, stream>>>(tgt_h, sa_in_wh + 2 * 256 * 256, sa_in_b + 512, nullptr, vb, nullptr, ROWS, 256, 256, 0);
  // 5) self-attention -> f16
  k_attn<<<(BS * NHEAD * NQ) / 8, blk, 0, stream>>>(qb, kb, vb, sa_h);
  // 6) sa output projection
  k_gemm<<<g256, blk, 0, stream>>>(sa_h, sa_out_wh, sa_out_b, nullptr, sa2, nullptr, ROWS, 256, 256, 0);
  // 7) tgt2 = LN(tgt + sa2)
  k_res_ln<<<ROWS, blk, 0, stream>>>(tgt, sa2, ln2_g, ln2_b, tgt2, nullptr);
  // 8) value = src @ val_w^T + val_b (fp32, gathered by MSDA)
  k_gemm<<<gval, blk, 0, stream>>>(src_h, val_wh, val_b, nullptr, value, nullptr, BS * NV, 256, 256, 0);
  // 9) qk2 = tgt2 + query_pos (f16)
  k_add_h<<<(int)(R / 4 / 256), blk, 0, stream>>>((const float4*)tgt2, (const float4*)qpos, (v4h*)qk_h, (int)(R / 4));
  // 10-11) sampling offsets + attention weights
  k_gemm<<<g256, blk, 0, stream>>>(qk_h, off_wh, off_b, nullptr, offr, nullptr, ROWS, 256, 256, 0);
  k_gemm<<<g128, blk, 0, stream>>>(qk_h, aw_wh,  aw_b,  nullptr, awr,  nullptr, ROWS, 128, 256, 0);
  // 12) deformable sampling -> f16
  k_msda<<<(BS * NQ * NHEAD) / 8, blk, 0, stream>>>(offr, awr, ref, value, msda_h);
  // 13) ca = msda @ out_w^T + out_b + tgt2
  k_gemm<<<g256, blk, 0, stream>>>(msda_h, out_wh, out_b, tgt2, ca, nullptr, ROWS, 256, 256, 0);
  // 14) tgt3 = LN(tgt2 + ca) (fp32 + f16 copy for FFN1)
  k_res_ln<<<ROWS, blk, 0, stream>>>(tgt2, ca, ln1_g, ln1_b, tgt3, tgt3_h);
  // 15) ffnh = relu(tgt3 @ ffn1^T + b) -> f16 only
  k_gemm<<<g1024, blk, 0, stream>>>(tgt3_h, ffn1_wh, ffn1_b, nullptr, nullptr, ffnh_h, ROWS, 1024, 256, 1);
  // 16) ffno = ffnh @ ffn2^T + b
  k_gemm<<<g256, blk, 0, stream>>>(ffnh_h, ffn2_wh, ffn2_b, nullptr, ffno, nullptr, ROWS, 256, 1024, 0);
  // 17) out = LN(tgt3 + ffno)
  k_res_ln<<<ROWS, blk, 0, stream>>>(tgt3, ffno, ln3_g, ln3_b, (float*)d_out, nullptr);
}
